// MultiHead_36567351558874
// MI455X (gfx1250) — matrix operand order repaired and checked by reading, not compiled, not run on hardware
//
#include <hip/hip_runtime.h>
#include <stdint.h>

// MI455X / gfx1250, wave32. bf16 WMMA 16x16x32, fp32 accumulate.
// Async global->LDS staging (ASYNCcnt) for W and K/V tiles, double buffered.
// d_ws requirement: (4M + 3M + 3*4M) bf16 elems = 38 MB.

typedef __attribute__((ext_vector_type(16))) __bf16 v16bf;
typedef __attribute__((ext_vector_type(8)))  __bf16 v8bf;
typedef __attribute__((ext_vector_type(8)))  float  v8f;

#define S_LEN 2048
#define EMB   1024
#define NHEAD 16
#define DHEAD 64

static __device__ __forceinline__ v8f wmma_bf16(v16bf a, v16bf b, v8f c) {
  return __builtin_amdgcn_wmma_f32_16x16x32_bf16(false, a, false, b, (short)0, c,
                                                 false, false);
}

static __device__ __forceinline__ v16bf pack2(v8bf lo, v8bf hi) {
  v16bf r;
#pragma unroll
  for (int i = 0; i < 8; ++i) { r[i] = lo[i]; r[8 + i] = hi[i]; }
  return r;
}

static __device__ __forceinline__ float rowmax16(float v) {
  v = fmaxf(v, __shfl_xor(v, 1, 32));
  v = fmaxf(v, __shfl_xor(v, 2, 32));
  v = fmaxf(v, __shfl_xor(v, 4, 32));
  v = fmaxf(v, __shfl_xor(v, 8, 32));
  return v;
}
static __device__ __forceinline__ float rowsum16(float v) {
  v += __shfl_xor(v, 1, 32);
  v += __shfl_xor(v, 2, 32);
  v += __shfl_xor(v, 4, 32);
  v += __shfl_xor(v, 8, 32);
  return v;
}

// Low 32 bits of a generic pointer to __shared__ = LDS byte address.
static __device__ __forceinline__ uint32_t lds_off(const void* p) {
  return (uint32_t)(uintptr_t)p;
}

// 16B async copy global -> LDS (VDST = per-lane LDS addr, VADDR = 64b global).
static __device__ __forceinline__ void cp_async16(uint32_t ldst, const void* gsrc) {
  asm volatile("global_load_async_to_lds_b128 %0, %1, off"
               :: "v"(ldst), "v"((uint64_t)(uintptr_t)gsrc)
               : "memory");
}

template <int N>
static __device__ __forceinline__ void wait_asynccnt() {
  asm volatile("s_wait_asynccnt %0" :: "i"(N) : "memory");
}

// ---------------------------------------------------------------------------
// Kernel 0: fp32 -> bf16 bulk convert (one-time; removes v_cvt from GEMM loop)
// ---------------------------------------------------------------------------
__global__ __launch_bounds__(256) void cvt_bf16_kernel(
    const float* __restrict__ src, unsigned short* __restrict__ dst) {
  const size_t i = ((size_t)blockIdx.x * 256 + threadIdx.x) * 8;
  v8f v = *(const v8f*)(src + i);
  v8bf o;
#pragma unroll
  for (int j = 0; j < 8; ++j) o[j] = (__bf16)v[j];
  *(v8bf*)((__bf16*)dst + i) = o;
}

// ---------------------------------------------------------------------------
// Kernel 1: fused QKV projection, y = x @ W^T + b (bf16 in, bf16 out).
// Block = 128 thr (4 waves); wave tile 64x64 (16 accumulators); k-step 32.
// W k-slice (64 rows x 32 k = 4KB) async-staged to double-buffered LDS and
// shared by all 4 waves (4x less W traffic than per-wave streaming).
//   z==0 -> Q [B,H,S,Dh]; z==1 -> K [B,H,S,Dh]; z==2 -> V^T [B,H,Dh,S]
// ---------------------------------------------------------------------------
__global__ __launch_bounds__(128) void qkv_proj_kernel(
    const unsigned short* __restrict__ xbf,
    const unsigned short* __restrict__ wbf,   // 3 * EMB*EMB bf16, z-major
    const float* __restrict__ bq, const float* __restrict__ bk,
    const float* __restrict__ bv,
    unsigned short* __restrict__ qout,
    unsigned short* __restrict__ kout,
    unsigned short* __restrict__ vtout) {
  __shared__ __bf16 wlds[2][64 * 32];

  const int z = blockIdx.z;
  const __bf16* W = (const __bf16*)wbf + (size_t)z * EMB * EMB;
  const float* bias = (z == 0) ? bq : (z == 1) ? bk : bv;

  const int tid  = threadIdx.x;
  const int lane = tid & 31;
  const int wid  = tid >> 5;
  const int col  = lane & 15;
  const int half = lane >> 4;
  const int m0 = blockIdx.x * 256 + wid * 64;  // 64 token rows per wave
  const int n0 = blockIdx.y * 64;              // 64 features per block

  auto stage_w = [&](int buf, int k) {
    const uint32_t base = lds_off(&wlds[buf][0]);
#pragma unroll
    for (int j = 0; j < 2; ++j) {
      const int id = tid + j * 128;       // 256 x 16B chunks
      const int row = id >> 2, part = id & 3;
      cp_async16(base + row * 64 + part * 16,
                 W + (size_t)(n0 + row) * EMB + k + part * 8);
    }
  };

  v8f acc[16];
#pragma unroll
  for (int i = 0; i < 16; ++i) acc[i] = v8f{};

  const __bf16* X = (const __bf16*)xbf;
  stage_w(0, 0);

  for (int kst = 0; kst < EMB / 32; ++kst) {
    const int k = kst * 32;
    const int buf = kst & 1;
    if (kst + 1 < EMB / 32) { stage_w(buf ^ 1, k + 32); wait_asynccnt<2>(); }
    else                    { wait_asynccnt<0>(); }
    __syncthreads();

    // A fragments: 4 x (16 rows x 32 k) of x, contiguous bf16 loads.
    v16bf a[4];
#pragma unroll
    for (int mi = 0; mi < 4; ++mi) {
      const __bf16* xr = X + (size_t)(m0 + mi * 16 + col) * EMB + k;
      a[mi] = pack2(*(const v8bf*)(xr + half * 16),
                    *(const v8bf*)(xr + half * 16 + 8));
    }
    // B fragments from LDS, 16 WMMAs.
#pragma unroll
    for (int ni = 0; ni < 4; ++ni) {
      v16bf bfrag = *(const v16bf*)(&wlds[buf][(ni * 16 + col) * 32 + half * 16]);
#pragma unroll
      for (int mi = 0; mi < 4; ++mi)
        acc[mi * 4 + ni] = wmma_bf16(a[mi], bfrag, acc[mi * 4 + ni]);
    }
    __syncthreads();  // buffer 'buf' is overwritten by next iteration's issue
  }

  // Epilogue: bias + bf16 stores (m0 multiple of 64 -> one batch per wave).
  const int b = m0 >> 11;
#pragma unroll
  for (int ni = 0; ni < 4; ++ni) {
    const int n = n0 + ni * 16 + col;
    const float bb = bias[n];
    const int h = n >> 6;
    const int d = n & 63;
#pragma unroll
    for (int mi = 0; mi < 4; ++mi) {
      const int s0 = (m0 & (S_LEN - 1)) + mi * 16 + 8 * half;
      if (z < 2) {
        __bf16* o = (__bf16*)((z == 0) ? qout : kout);
        size_t base = (((size_t)b * NHEAD + h) * S_LEN + s0) * DHEAD + d;
#pragma unroll
        for (int r = 0; r < 8; ++r)
          o[base + (size_t)r * DHEAD] = (__bf16)(acc[mi * 4 + ni][r] + bb);
      } else {
        __bf16* o = (__bf16*)vtout +
                    (((size_t)b * NHEAD + h) * DHEAD + d) * S_LEN + s0;
        v8bf pk;
#pragma unroll
        for (int r = 0; r < 8; ++r) pk[r] = (__bf16)(acc[mi * 4 + ni][r] + bb);
        *(v8bf*)o = pk;
      }
    }
  }
}

// ---------------------------------------------------------------------------
// Kernel 2: flash attention. Block = 128 thr (4 waves); wave owns 16 q rows;
// kv chunk = 32. K chunk (32x64 bf16, contiguous) and V^T slice (64x32) are
// async-staged once per block into double-buffered LDS (all 4 waves read the
// SAME fragments). P goes C-layout -> A-layout via per-wave LDS + s_wait_dscnt.
// ---------------------------------------------------------------------------
__global__ __launch_bounds__(128) void attn_kernel(
    const unsigned short* __restrict__ Qbf,
    const unsigned short* __restrict__ Kbf,
    const unsigned short* __restrict__ Vt,
    float* __restrict__ out) {
  __shared__ __bf16 kvbuf[2][2048 + 2048];  // K[32][64] then V[64][32]
  __shared__ __bf16 plds[4][16 * 32];

  const int tid  = threadIdx.x;
  const int lane = tid & 31;
  const int wid  = tid >> 5;
  const int col  = lane & 15;
  const int half = lane >> 4;
  const int h  = blockIdx.y;
  const int b  = blockIdx.z;
  const int q0 = blockIdx.x * 64 + wid * 16;

  const __bf16* Q  = (const __bf16*)Qbf + ((size_t)b * NHEAD + h) * S_LEN * DHEAD;
  const __bf16* Kg = (const __bf16*)Kbf + ((size_t)b * NHEAD + h) * S_LEN * DHEAD;
  const __bf16* Vg = (const __bf16*)Vt  + ((size_t)b * NHEAD + h) * DHEAD * S_LEN;

  auto stage_kv = [&](int buf, int c) {
    const uint32_t kbase = lds_off(&kvbuf[buf][0]);
    const uint32_t vbase = lds_off(&kvbuf[buf][2048]);
#pragma unroll
    for (int j = 0; j < 2; ++j) {          // K: 4KB contiguous
      const int id = tid + j * 128;
      cp_async16(kbase + id * 16, Kg + (size_t)c * DHEAD + id * 8);
    }
#pragma unroll
    for (int j = 0; j < 2; ++j) {          // V^T: 64 rows x 64B
      const int id = tid + j * 128;
      const int row = id >> 2, part = id & 3;
      cp_async16(vbase + row * 64 + part * 16,
                 Vg + (size_t)row * S_LEN + c + part * 8);
    }
  };

  // Q A-fragments (d 0..31, 32..63), live for the whole kv loop.
  const __bf16* qrow = Q + (size_t)(q0 + col) * DHEAD;
  v16bf aq0 = pack2(*(const v8bf*)(qrow + half * 16),
                    *(const v8bf*)(qrow + half * 16 + 8));
  v16bf aq1 = pack2(*(const v8bf*)(qrow + 32 + half * 16),
                    *(const v8bf*)(qrow + 32 + half * 16 + 8));

  float mrow[8], lrow[8];
#pragma unroll
  for (int r = 0; r < 8; ++r) { mrow[r] = -1e30f; lrow[r] = 0.0f; }
  v8f o0{}, o1{}, o2{}, o3{};

  __bf16* my = &plds[wid][0];
  const float kscale = 0.125f;  // 1/sqrt(64)

  stage_kv(0, 0);

  for (int c = 0; c < S_LEN; c += 32) {
    const int buf = (c >> 5) & 1;
    if (c + 32 < S_LEN) { stage_kv(buf ^ 1, c + 32); wait_asynccnt<4>(); }
    else                { wait_asynccnt<0>(); }
    __syncthreads();

    const __bf16* Kl = &kvbuf[buf][0];
    const __bf16* Vl = &kvbuf[buf][2048];

    v16bf bk00 = *(const v16bf*)(Kl + (col)      * 64 + half * 16);
    v16bf bk01 = *(const v16bf*)(Kl + (col)      * 64 + 32 + half * 16);
    v16bf bk10 = *(const v16bf*)(Kl + (16 + col) * 64 + half * 16);
    v16bf bk11 = *(const v16bf*)(Kl + (16 + col) * 64 + 32 + half * 16);

    v8f s0{}, s1{};
    s0 = wmma_bf16(aq0, bk00, s0);
    s0 = wmma_bf16(aq1, bk01, s0);
    s1 = wmma_bf16(aq0, bk10, s1);
    s1 = wmma_bf16(aq1, bk11, s1);

    // Online softmax for this 32-wide chunk.
    float p0[8], p1[8], corr[8];
#pragma unroll
    for (int r = 0; r < 8; ++r) {
      float v0 = s0[r] * kscale;
      float v1 = s1[r] * kscale;
      float cm = rowmax16(fmaxf(v0, v1));
      float mn = fmaxf(mrow[r], cm);
      float cr = __expf(mrow[r] - mn);
      float e0 = __expf(v0 - mn);
      float e1 = __expf(v1 - mn);
      lrow[r] = lrow[r] * cr + rowsum16(e0 + e1);
      mrow[r] = mn;
      corr[r] = cr;
      p0[r] = e0;
      p1[r] = e1;
    }
#pragma unroll
    for (int r = 0; r < 8; ++r) {
      o0[r] *= corr[r]; o1[r] *= corr[r]; o2[r] *= corr[r]; o3[r] *= corr[r];
    }

    // C-layout -> A-layout for P via per-wave LDS tile (wave-internal).
#pragma unroll
    for (int r = 0; r < 8; ++r) {
      const int row = r + 8 * half;
      my[row * 32 + col]      = (__bf16)p0[r];
      my[row * 32 + col + 16] = (__bf16)p1[r];
    }
    asm volatile("s_wait_dscnt 0" ::: "memory");
    v16bf pa = pack2(*(const v8bf*)(my + col * 32 + half * 16),
                     *(const v8bf*)(my + col * 32 + half * 16 + 8));

    o0 = wmma_bf16(pa, *(const v16bf*)(Vl + (col)      * 32 + half * 16), o0);
    o1 = wmma_bf16(pa, *(const v16bf*)(Vl + (16 + col) * 32 + half * 16), o1);
    o2 = wmma_bf16(pa, *(const v16bf*)(Vl + (32 + col) * 32 + half * 16), o2);
    o3 = wmma_bf16(pa, *(const v16bf*)(Vl + (48 + col) * 32 + half * 16), o3);

    __syncthreads();  // protect 'buf' before next iteration overwrites it
  }

  // Epilogue: normalize and store fp32 [B,S,E].
  size_t obase = ((size_t)b * S_LEN + q0 + 8 * half) * EMB + h * DHEAD + col;
#pragma unroll
  for (int r = 0; r < 8; ++r) {
    const float inv = 1.0f / lrow[r];
    float* op = out + obase + (size_t)r * EMB;
    op[0]  = o0[r] * inv;
    op[16] = o1[r] * inv;
    op[32] = o2[r] * inv;
    op[48] = o3[r] * inv;
  }
}

extern "C" void kernel_launch(void* const* d_in, const int* in_sizes, int n_in,
                              void* d_out, int out_size, void* d_ws,
                              size_t ws_size, hipStream_t stream) {
  const float* x  = (const float*)d_in[0];
  const float* Wq = (const float*)d_in[1];
  const float* bq = (const float*)d_in[2];
  const float* Wk = (const float*)d_in[3];
  const float* bk = (const float*)d_in[4];
  const float* Wv = (const float*)d_in[5];
  const float* bv = (const float*)d_in[6];
  float* out = (float*)d_out;

  const size_t NX = (size_t)2 * S_LEN * EMB;   // 4M
  const size_t NW = (size_t)EMB * EMB;         // 1M
  const size_t NQ = (size_t)2 * NHEAD * S_LEN * DHEAD;  // 4M

  unsigned short* xbf = (unsigned short*)d_ws;
  unsigned short* wbf = xbf + NX;        // 3*NW
  unsigned short* qbf = wbf + 3 * NW;
  unsigned short* kbf = qbf + NQ;
  unsigned short* vtb = kbf + NQ;

  // One-time fp32 -> bf16 conversion (2048 elems per block).
  cvt_bf16_kernel<<<(unsigned)(NX / 2048), 256, 0, stream>>>(x, xbf);
  cvt_bf16_kernel<<<(unsigned)(NW / 2048), 256, 0, stream>>>(Wq, wbf);
  cvt_bf16_kernel<<<(unsigned)(NW / 2048), 256, 0, stream>>>(Wk, wbf + NW);
  cvt_bf16_kernel<<<(unsigned)(NW / 2048), 256, 0, stream>>>(Wv, wbf + 2 * NW);

  // QKV projection: block tile 256(M) x 64(N), z = {Q,K,V}.
  dim3 pgrid(4096 / 256, EMB / 64, 3);
  qkv_proj_kernel<<<pgrid, 128, 0, stream>>>(xbf, wbf, bq, bk, bv,
                                             qbf, kbf, vtb);

  // Flash attention: 64 query rows per block, per (head, batch).
  dim3 agrid(S_LEN / 64, NHEAD, 2);
  attn_kernel<<<agrid, 128, 0, stream>>>(qbf, kbf, vtb, out);
}
